// GATMHA_89472758710696
// MI455X (gfx1250) — matrix operand order repaired and checked by reading, not compiled, not run on hardware
//
#include <hip/hip_runtime.h>

typedef __attribute__((ext_vector_type(2)))  float    v2f;
typedef __attribute__((ext_vector_type(8)))  float    v8f;
typedef __attribute__((ext_vector_type(8)))  _Float16 v8h;
typedef __attribute__((ext_vector_type(16))) _Float16 v16h;

#define SLOPE 0.2f
#define B_  8
#define N_  1024
#define DI  128
#define H_  8
#define HD  16

// ---------------------------------------------------------------------------
// Kernel 1: per-head projection g = h @ W[head]  (f32 WMMA 16x16x4),
// plus e_src = g . a_src, e_dst = g . a_dst, and transposed f16 store of g.
// grid = H_ * 64 blocks, 256 threads (8 waves, 16 rows per wave).
// ---------------------------------------------------------------------------
__global__ __launch_bounds__(256)
void gat_proj_kernel(const float* __restrict__ hIn,
                     const float* __restrict__ Wg,
                     const float* __restrict__ Wa,
                     _Float16* __restrict__ gT,
                     float* __restrict__ eSrc,
                     float* __restrict__ eDst)
{
    __shared__ float tile[8][16][17];

    const int head     = blockIdx.x >> 6;   // 8 heads
    const int rowGroup = blockIdx.x & 63;   // 64 groups of 128 rows (of 8192)
    const int wave = threadIdx.x >> 5;
    const int lane = threadIdx.x & 31;
    const int m    = lane & 15;
    const int hi   = lane >> 4;

    const int row0 = rowGroup * 128 + wave * 16;       // global row in [0,8192)
    const float* hrow = hIn + (size_t)(row0 + m) * DI;
    const float* Wh   = Wg  + (size_t)head * DI * HD;

    // f32 WMMA: A 16x4 (lane: M=m, VGPR p: K = p + 2*hi)
    //           B 4x16 (lane: N=m, VGPR p: K = p + 2*hi)
    v8f acc = {};
    #pragma unroll 4
    for (int k0 = 0; k0 < DI; k0 += 4) {
        const int k = k0 + 2 * hi;
        v2f a = *(const v2f*)(hrow + k);
        v2f b;
        b.x = Wh[(k + 0) * HD + m];
        b.y = Wh[(k + 1) * HD + m];
        acc = __builtin_amdgcn_wmma_f32_16x16x4_f32(false, a, false, b,
                                                    (short)0, acc, false, false);
    }

    // D layout: VGPR r -> row (r + 8*hi), col m
    #pragma unroll
    for (int r = 0; r < 8; ++r)
        tile[wave][r + 8 * hi][m] = acc[r];
    __syncthreads();

    const int bIdx = row0 >> 10;
    const int nIdx = row0 & 1023;
    const int hb   = head * B_ + bIdx;

    if (lane < 16) {
        float s0 = 0.f, s1 = 0.f;
        #pragma unroll
        for (int d = 0; d < HD; ++d) {
            const float g = tile[wave][lane][d];
            s0 += g * Wa[head * 2 * HD + d];
            s1 += g * Wa[head * 2 * HD + HD + d];
        }
        eSrc[(size_t)hb * N_ + nIdx + lane] = s0;
        eDst[(size_t)hb * N_ + nIdx + lane] = s1;
    }

    // Transposed f16 store: lane handles d = m, rows 8*hi .. 8*hi+7
    v8h pack;
    #pragma unroll
    for (int q = 0; q < 8; ++q)
        pack[q] = (_Float16)tile[wave][8 * hi + q][m];
    *(v8h*)(gT + ((size_t)hb * HD + m) * N_ + nIdx + 8 * hi) = pack;
}

// ---------------------------------------------------------------------------
// Kernel 2: attention + output.  s_ij = max(E1[j], t_i * E2[j]) (exact
// rewrite of exp(leakyrelu(e_i+e_j) - rowmax)).  attn@g and row sums via
// v_wmma_f32_16x16x32_f16 (ones-matrix B for sums).  Score generation uses
// packed f16 ops (v_pk_mul_f16 / v_pk_max_num_f16).
// grid = 64 hb * 4 row-quads, 256 threads; each wave: two 16-row tiles.
// ---------------------------------------------------------------------------
__global__ __launch_bounds__(256)
void gat_attn_kernel(const _Float16* __restrict__ gT,
                     const float* __restrict__ eSrc,
                     const float* __restrict__ eDst,
                     float* __restrict__ out)
{
    __shared__ __align__(16) _Float16 gS[HD * N_];   // 32 KB  [d][j]
    __shared__ __align__(16) _Float16 E1[N_];        // 2 KB
    __shared__ __align__(16) _Float16 E2[N_];        // 2 KB
    __shared__ float red[8];

    const int hb   = blockIdx.x >> 2;   // 64 (head,batch) pairs
    const int quad = blockIdx.x & 3;    // 4 quads of 256 rows
    const int head = hb >> 3;
    const int bIdx = hb & 7;
    const int tid  = threadIdx.x;
    const int wave = tid >> 5;
    const int lane = tid & 31;

    // stage gT -> LDS (32 KB)
    {
        const float4* src = (const float4*)(gT + (size_t)hb * HD * N_);
        float4* dst = (float4*)gS;
        #pragma unroll
        for (int it = 0; it < 8; ++it)
            dst[tid + 256 * it] = src[tid + 256 * it];
    }

    // M = max_j e_dst[j]
    const float* ed = eDst + (size_t)hb * N_;
    float mx = -3.4e38f;
    #pragma unroll
    for (int it = 0; it < 4; ++it)
        mx = fmaxf(mx, ed[tid + 256 * it]);
    #pragma unroll
    for (int off = 16; off >= 1; off >>= 1)
        mx = fmaxf(mx, __shfl_xor(mx, off, 32));
    if (lane == 0) red[wave] = mx;
    __syncthreads();
    mx = red[0];
    #pragma unroll
    for (int wv = 1; wv < 8; ++wv) mx = fmaxf(mx, red[wv]);

    // E1/E2 tables
    #pragma unroll
    for (int it = 0; it < 4; ++it) {
        const int j = tid + 256 * it;
        const float d = ed[j] - mx;
        E1[j] = (_Float16)__expf(d);
        E2[j] = (_Float16)__expf(SLOPE * d);
    }
    __syncthreads();

    const int mrow = lane & 15;   // A row / B col / D col
    const int hi   = lane >> 4;

    v16h ones;
    #pragma unroll
    for (int q = 0; q < 16; ++q) ones[q] = (_Float16)1.0f;

    for (int tt = 0; tt < 2; ++tt) {
        const int tileI = quad * 16 + wave * 2 + tt;
        const int i0 = tileI * 16;

        const float es = eSrc[(size_t)hb * N_ + i0 + mrow];
        const _Float16 th = (_Float16)__expf(-0.8f * fmaxf(es + mx, 0.f));

        v8f co = {}, cs = {};
        for (int c = 0; c < 32; ++c) {
            // A f16 16x32 layout: e<8 -> k = 8*hi + e ; e>=8 -> k = 16 + 8*hi + (e-8)
            const int b1 = c * 32 + hi * 8;
            const int b2 = b1 + 16;
            v8h e1a = *(const v8h*)(E1 + b1);
            v8h e1b = *(const v8h*)(E1 + b2);
            v8h e2a = *(const v8h*)(E2 + b1);
            v8h e2b = *(const v8h*)(E2 + b2);

            // packed: v_pk_mul_f16 + v_pk_max_num_f16
            v8h s1 = __builtin_elementwise_max(e1a, e2a * th);
            v8h s2 = __builtin_elementwise_max(e1b, e2b * th);
            v16h A = __builtin_shufflevector(s1, s2,
                        0, 1, 2, 3, 4, 5, 6, 7, 8, 9, 10, 11, 12, 13, 14, 15);

            // B f16 32x16 layout: e<8 -> K = 8*hi + e ; e>=8 -> K = 16 + 8*hi + (e-8)
            const v8h bga = *(const v8h*)(gS + (size_t)mrow * N_ + b1);
            const v8h bgb = *(const v8h*)(gS + (size_t)mrow * N_ + b2);
            const v16h Bg = __builtin_shufflevector(bga, bgb,
                        0, 1, 2, 3, 4, 5, 6, 7, 8, 9, 10, 11, 12, 13, 14, 15);

            co = __builtin_amdgcn_wmma_f32_16x16x32_f16(false, A, false, Bg,
                                                        (short)0, co, false, false);
            cs = __builtin_amdgcn_wmma_f32_16x16x32_f16(false, A, false, ones,
                                                        (short)0, cs, false, false);
        }

        // D layout: VGPR r -> row i0 + r + 8*hi, col mrow; cs holds row sums
        float* obase = out + (size_t)bIdx * N_ * (H_ * HD) + head * HD + mrow;
        #pragma unroll
        for (int r = 0; r < 8; ++r) {
            const int i = i0 + r + 8 * hi;
            const float v = co[r] * __builtin_amdgcn_rcpf(cs[r]);
            obase[(size_t)i * (H_ * HD)] = fmaxf(v, 0.f);
        }
    }
}

// ---------------------------------------------------------------------------
extern "C" void kernel_launch(void* const* d_in, const int* in_sizes, int n_in,
                              void* d_out, int out_size, void* d_ws, size_t ws_size,
                              hipStream_t stream) {
    (void)in_sizes; (void)n_in; (void)out_size; (void)ws_size;
    const float* h  = (const float*)d_in[0];
    const float* W  = (const float*)d_in[1];
    const float* Wa = (const float*)d_in[2];
    float* out = (float*)d_out;

    _Float16* gT = (_Float16*)d_ws;                                  // 2 MB
    float* eSrc  = (float*)((char*)d_ws + (size_t)H_ * B_ * HD * N_ * sizeof(_Float16));
    float* eDst  = eSrc + (size_t)H_ * B_ * N_;                      // +256 KB each

    gat_proj_kernel<<<dim3(H_ * 64), dim3(256), 0, stream>>>(h, W, Wa, gT, eSrc, eDst);
    gat_attn_kernel<<<dim3(H_ * B_ * 4), dim3(256), 0, stream>>>(gT, eSrc, eDst, out);
}
